// LlamaFlashAttentionMasked_59176059405005
// MI455X (gfx1250) — matrix operand order repaired and checked by reading, not compiled, not run on hardware
//
#include <hip/hip_runtime.h>
#include <hip/hip_bf16.h>

typedef __attribute__((ext_vector_type(16))) __bf16 v16bf;
typedef __attribute__((ext_vector_type(8)))  __bf16 v8bf;
typedef __attribute__((ext_vector_type(8)))  float  v8f;
typedef __attribute__((ext_vector_type(4)))  unsigned int u32x4;
typedef __attribute__((ext_vector_type(8)))  int i32x8;
typedef __attribute__((ext_vector_type(4)))  int i32x4;

#define HIDDEN 4096
#define NH     32
#define NKV    8
#define HD     128
#define LCK    3
#define BB     2
#define SS     1024
#define MROWS  (BB*SS)          // 2048
#define NEGBIG (-1e30f)

static __device__ __forceinline__ v8f wmma_bf16(v16bf a, v16bf b, v8f c) {
  return __builtin_amdgcn_wmma_f32_16x16x32_bf16(false, a, false, b, (short)0, c, false, false);
}

static __device__ __forceinline__ v16bf cat8(v8bf lo, v8bf hi) {
  return __builtin_shufflevector(lo, hi, 0,1,2,3,4,5,6,7,8,9,10,11,12,13,14,15);
}

// ---------------------------------------------------------------- TDM -------
// 2D bf16 tile load Global -> LDS via Tensor Data Mover (6-arg builtin lane).
// tile_d0 elements along contiguous dim (row), tile_d1 rows, row stride
// stride0 (elements). LDS pad per D# spec: interval code v -> 2^(v+1) DWORDs
// stored between pads; amount code v -> (v+1) DWORDs inserted.
static __device__ __forceinline__ void tdm_load_2d_bf16(
    const __bf16* gptr, unsigned lds_off,
    unsigned tile_d0, unsigned tile_d1, unsigned stride0,
    unsigned pad_int_code, unsigned pad_amt_code, unsigned pad_en) {
  unsigned long long ga = (unsigned long long)(uintptr_t)gptr;
  u32x4 g0;
  g0[0] = 1u;                                   // count=1 (valid), user mode
  g0[1] = lds_off;                              // lds_addr (bytes)
  g0[2] = (unsigned)(ga & 0xffffffffu);         // global_addr[31:0]
  g0[3] = (unsigned)((ga >> 32) & 0x01ffffffu)  // global_addr[56:32]
        | (2u << 30);                           // type=2 ("image")
  i32x8 g1;
  g1[0] = (int)((1u << 16)                      // data_size=1 -> 2 bytes
        | (pad_en << 20)                        // pad_enable
        | (pad_int_code << 22)                  // pad_interval
        | (pad_amt_code << 25));                // pad_amount
  g1[1] = (int)((tile_d0 & 0xffffu) << 16);     // tensor_dim0[15:0] (== tile)
  g1[2] = (int)(((tile_d0 >> 16) & 0xffffu) | ((tile_d1 & 0xffffu) << 16));
  g1[3] = (int)(((tile_d1 >> 16) & 0xffffu) | ((tile_d0 & 0xffffu) << 16)); // tile_dim0
  g1[4] = (int)(tile_d1 & 0xffffu);             // tile_dim1, tile_dim2=0
  g1[5] = (int)stride0;                         // tensor_dim0_stride[31:0]
  g1[6] = 0;                                    // stride0[47:32]=0, stride1=0
  g1[7] = 0;
  i32x4 z4 = {0, 0, 0, 0};
  i32x8 z8 = {0, 0, 0, 0, 0, 0, 0, 0};
  __builtin_amdgcn_tensor_load_to_lds(g0, g1, z4, z4, z8, 0);
}

// ---------------------------------------------------------------- convert ---
__global__ void f32_to_bf16_kernel(const float* __restrict__ in,
                                   __bf16* __restrict__ out, int n) {
  int i = blockIdx.x * blockDim.x + threadIdx.x;
  if (i < n) out[i] = (__bf16)in[i];
}

// ------------------------------------------------------------------- GEMM ---
// C[M,N] = A[M,K](bf16) * Bw[K,N](fp32, converted to bf16 on LDS store)
// block tile 128x128, K-step 32, 256 threads = 8 waves (4x2), wave tile 32x64.
// Double-buffered: TDM DMA for tile i+1 + B-convert run under tile-i WMMAs.
#define GT_M 128
#define GT_N 128
#define GT_K 32
#define LDT  40   // LDS pitch in halves (padded; matches TDM pad 64B/16B)

__launch_bounds__(256)
__global__ void gemm_bf16_kernel(const __bf16* __restrict__ A,
                                 const float* __restrict__ Bw,
                                 float* __restrict__ C,
                                 int M, int N, int K) {
  __shared__ __align__(16) __bf16 Asb[2][GT_M * LDT];
  __shared__ __align__(16) __bf16 Bsb[2][GT_N * LDT];   // transposed: [n][k]

  const int t    = threadIdx.x;
  const int lane = t & 31;
  const int w    = t >> 5;
  const int wm   = w >> 1;            // 0..3
  const int wn   = w & 1;             // 0..1
  const int ml   = lane & 15;
  const int hf   = lane >> 4;
  const int m0   = blockIdx.y * GT_M;
  const int n0   = blockIdx.x * GT_N;

  // B-tile loader mapping: 2 consecutive k x 8 n per thread
  const int kp = (t & 15) * 2;
  const int n8 = (t >> 4) * 8;

  const unsigned lds_as0 = (unsigned)(uintptr_t)(&Asb[0][0]);
  const unsigned lds_as1 = (unsigned)(uintptr_t)(&Asb[1][0]);

  auto stage_b = [&](int kk2, __bf16* bs) {
    const float4* r0 = (const float4*)(Bw + (size_t)(kk2 + kp) * N + n0 + n8);
    const float4* r1 = (const float4*)(Bw + (size_t)(kk2 + kp + 1) * N + n0 + n8);
    float4 a0 = r0[0], a1 = r0[1];
    float4 b0 = r1[0], b1 = r1[1];
    float lo[8] = {a0.x, a0.y, a0.z, a0.w, a1.x, a1.y, a1.z, a1.w};
    float hi[8] = {b0.x, b0.y, b0.z, b0.w, b1.x, b1.y, b1.z, b1.w};
#pragma unroll
    for (int i = 0; i < 8; i++) {
      union { __bf16 h[2]; unsigned u; } pk;
      pk.h[0] = (__bf16)lo[i];
      pk.h[1] = (__bf16)hi[i];
      *(unsigned*)(&bs[(n8 + i) * LDT + kp]) = pk.u;
    }
  };

  v8f acc[2][4];
#pragma unroll
  for (int i = 0; i < 2; i++)
#pragma unroll
    for (int j = 0; j < 4; j++)
#pragma unroll
      for (int e = 0; e < 8; e++) acc[i][j][e] = 0.0f;

  // prologue: stage tile 0 into buffer 0
  if (w == 0)
    tdm_load_2d_bf16(A + (size_t)m0 * K, lds_as0,
                     GT_K, GT_M, (unsigned)K, 3u, 3u, 1u);
  stage_b(0, &Bsb[0][0]);

  const int nk = K / GT_K;
  for (int i = 0; i < nk; i++) {
    const int cur = i & 1;
    const int nxt = cur ^ 1;
    if (w == 0) __builtin_amdgcn_s_wait_tensorcnt(0);
    __syncthreads();

    const __bf16* As = &Asb[cur][0];
    const __bf16* Bs = &Bsb[cur][0];
    v16bf af[2], bfrag[4];
#pragma unroll
    for (int mi = 0; mi < 2; mi++) {
      const __bf16* p = &As[(wm * 32 + mi * 16 + ml) * LDT + 8 * hf];
      af[mi] = cat8(*(const v8bf*)p, *(const v8bf*)(p + 16));
    }
#pragma unroll
    for (int ni = 0; ni < 4; ni++) {
      const __bf16* p = &Bs[(wn * 64 + ni * 16 + ml) * LDT + 8 * hf];
      bfrag[ni] = cat8(*(const v8bf*)p, *(const v8bf*)(p + 16));
    }

    // launch next tile's DMA + B conversion under this tile's WMMAs
    if (i + 1 < nk) {
      if (w == 0)
        tdm_load_2d_bf16(A + (size_t)m0 * K + (size_t)(i + 1) * GT_K,
                         (nxt ? lds_as1 : lds_as0),
                         GT_K, GT_M, (unsigned)K, 3u, 3u, 1u);
      stage_b((i + 1) * GT_K, &Bsb[nxt][0]);
      __builtin_prefetch(Bw + (size_t)((i + 1) * GT_K + kp) * N + n0 + n8, 0, 1);
    }

#pragma unroll
    for (int mi = 0; mi < 2; mi++)
#pragma unroll
      for (int ni = 0; ni < 4; ni++)
        acc[mi][ni] = wmma_bf16(af[mi], bfrag[ni], acc[mi][ni]);
  }

  // epilogue: C layout VGPR r -> row r + 8*hf, col = ml
#pragma unroll
  for (int mi = 0; mi < 2; mi++) {
#pragma unroll
    for (int ni = 0; ni < 4; ni++) {
      const int rowb = m0 + wm * 32 + mi * 16 + 8 * hf;
      const int col  = n0 + wn * 64 + ni * 16 + ml;
#pragma unroll
      for (int r = 0; r < 8; r++)
        C[(size_t)(rowb + r) * N + col] = acc[mi][ni][r];
    }
  }
}

// ------------------------------------------------------------------- RoPE ---
// X: [MROWS*nheads, 128] fp32 -> Y bf16 with RoPE applied (pos = row index % S)
__global__ void rope_bf16_kernel(const float* __restrict__ X,
                                 __bf16* __restrict__ Y, int nheads) {
  int idx = blockIdx.x * blockDim.x + threadIdx.x;   // pair index
  int i   = idx & 63;
  int row = idx >> 6;                                // (b*S + s)*nheads + h
  int pos = (row / nheads) % SS;
  size_t base = (size_t)row * HD;
  float fr = (float)pos * __expf(-(float)i * 0.14391565298394886f);
  float c = cosf(fr), s = sinf(fr);
  float x0 = X[base + i], x1 = X[base + i + 64];
  Y[base + i]      = (__bf16)(x0 * c - x1 * s);
  Y[base + i + 64] = (__bf16)(x1 * c + x0 * s);
}

// ------------------------------------------------- V transpose + convert ----
// V: [B,S,NKV,HD] fp32 -> Vt: [B,NKV,HD,S] bf16
__global__ void vtrans_kernel(const float* __restrict__ V,
                              __bf16* __restrict__ Vt) {
  int idx = blockIdx.x * blockDim.x + threadIdx.x;
  int s  = idx & (SS - 1);
  int d  = (idx >> 10) & (HD - 1);
  int hk = (idx >> 17) & (NKV - 1);
  int b  = idx >> 20;
  Vt[idx] = (__bf16)V[((size_t)((b * SS + s) * NKV + hk)) * HD + d];
}

// -------------------------------------------------------- flash attention ---
// grid (S/64, NH, B), block 128 (4 waves); wave owns 16 q-rows.
// K/V tiles for each 32-wide KV chunk are staged once per block via TDM.
#define KPITCH 136   // K tile pitch in halves (128 + 8 pad)

__launch_bounds__(128)
__global__ void attn_kernel(const __bf16* __restrict__ Qb,
                            const __bf16* __restrict__ Kb,
                            const __bf16* __restrict__ Vt,
                            const float* __restrict__ Ksuf,
                            const float* __restrict__ Vsuf,
                            const int* __restrict__ validp,
                            __bf16* __restrict__ Ob) {
  const int valid = validp[0];
  const int w    = threadIdx.x >> 5;
  const int lane = threadIdx.x & 31;
  const int n    = lane & 15;
  const int hf   = lane >> 4;
  const int b    = blockIdx.z;
  const int h    = blockIdx.y;
  const int hk   = h >> 2;
  const int qbase = blockIdx.x * 64 + w * 16;
  const float scale = 0.08838834764831845f;   // 1/sqrt(128)

  __shared__ __align__(16) __bf16 Kt[32 * KPITCH];   // [kv][d]
  __shared__ __align__(16) __bf16 Vs[128 * LDT];     // [d][kv]
  __shared__ __align__(16) __bf16 Ps[4][16 * LDT];
  __bf16* pw = Ps[w];
  const unsigned lds_kt = (unsigned)(uintptr_t)(&Kt[0]);
  const unsigned lds_vs = (unsigned)(uintptr_t)(&Vs[0]);

  // Q fragments (A layout), loaded once: lane row = qbase+n, K = d
  v16bf qf[4];
  {
    const __bf16* qrow = Qb + ((size_t)((b * SS + (qbase + n)) * NH + h)) * HD;
#pragma unroll
    for (int dc = 0; dc < 4; dc++) {
      const __bf16* p = qrow + dc * 32 + 8 * hf;
      qf[dc] = cat8(*(const v8bf*)p, *(const v8bf*)(p + 16));
    }
  }

  v8f   Oacc[8];
  float mrow[8], lrow[8];
#pragma unroll
  for (int t = 0; t < 8; t++)
#pragma unroll
    for (int e = 0; e < 8; e++) Oacc[t][e] = 0.0f;
#pragma unroll
  for (int r = 0; r < 8; r++) { mrow[r] = -3.0e38f; lrow[r] = 0.0f; }

  const int nchunks = (blockIdx.x * 64 + 64) >> 5;   // uniform across block
  for (int c = 0; c < nchunks; c++) {
    const int kv0 = c * 32;

    // ---- stage K tile (32 kv x 128 d) and V^T tile (128 d x 32 kv) ----
    if (w == 0) {
      tdm_load_2d_bf16(Kb + ((size_t)((b * SS + kv0) * NKV + hk)) * HD, lds_kt,
                       /*tile_d0=*/HD, /*tile_d1=*/32,
                       /*stride0=*/(unsigned)(NKV * HD),
                       /*pad: 256B interval, 16B amount -> pitch 136*/ 5u, 3u, 1u);
      tdm_load_2d_bf16(Vt + ((size_t)(b * NKV + hk) * HD) * SS + kv0, lds_vs,
                       /*tile_d0=*/32, /*tile_d1=*/HD,
                       /*stride0=*/(unsigned)SS,
                       /*pad: 64B interval, 16B amount -> pitch 40*/ 3u, 3u, 1u);
      __builtin_amdgcn_s_wait_tensorcnt(0);
    }
    __syncthreads();

    // ---- scores: two 16x16 tiles over 4 d-chunks (K from LDS) ----
    v8f s0, s1;
#pragma unroll
    for (int e = 0; e < 8; e++) { s0[e] = 0.0f; s1[e] = 0.0f; }
#pragma unroll
    for (int dc = 0; dc < 4; dc++) {
      const __bf16* k0p = &Kt[n * KPITCH + dc * 32 + 8 * hf];
      const __bf16* k1p = &Kt[(16 + n) * KPITCH + dc * 32 + 8 * hf];
      v16bf kf0 = cat8(*(const v8bf*)k0p, *(const v8bf*)(k0p + 16));
      v16bf kf1 = cat8(*(const v8bf*)k1p, *(const v8bf*)(k1p + 16));
      s0 = wmma_bf16(qf[dc], kf0, s0);
      s1 = wmma_bf16(qf[dc], kf1, s1);
    }

    // ---- mask + online softmax update ----
#pragma unroll
    for (int r = 0; r < 8; r++) {
      const int row = qbase + r + 8 * hf;
      const int c0  = kv0 + n, c1 = kv0 + 16 + n;
      float v0 = ((c0 <= row) && (c0 < valid)) ? s0[r] * scale : NEGBIG;
      float v1 = ((c1 <= row) && (c1 < valid)) ? s1[r] * scale : NEGBIG;
      float vm = fmaxf(v0, v1);
      for (int off = 8; off; off >>= 1) vm = fmaxf(vm, __shfl_xor(vm, off, 32));
      float mn = fmaxf(mrow[r], vm);
      float al = __expf(mrow[r] - mn);
      mrow[r] = mn;
      float p0 = __expf(v0 - mn);
      float p1 = __expf(v1 - mn);
      float rs = p0 + p1;
      for (int off = 8; off; off >>= 1) rs += __shfl_xor(rs, off, 32);
      lrow[r] = lrow[r] * al + rs;
#pragma unroll
      for (int t = 0; t < 8; t++) Oacc[t][r] *= al;
      pw[(r + 8 * hf) * LDT + n]      = (__bf16)p0;
      pw[(r + 8 * hf) * LDT + 16 + n] = (__bf16)p1;
    }
    __syncthreads();

    // ---- PV: P (A layout from LDS) x V^T tiles (from LDS) ----
    v16bf pf;
    {
      const __bf16* p = &pw[n * LDT + 8 * hf];
      pf = cat8(*(const v8bf*)p, *(const v8bf*)(p + 16));
    }
#pragma unroll
    for (int t = 0; t < 8; t++) {
      const __bf16* vp = &Vs[(t * 16 + n) * LDT + 8 * hf];
      v16bf vf = cat8(*(const v8bf*)vp, *(const v8bf*)(vp + 16));
      Oacc[t] = wmma_bf16(pf, vf, Oacc[t]);
    }
    __syncthreads();   // protect Kt/Vs/Ps before next chunk's staging
  }

  // ---- suffix diagonal logits: wave-cooperative dots, rows 0..15 ----
  float sdot[8][3];
#pragma unroll
  for (int rr = 0; rr < 16; rr++) {
    const int qg = qbase + rr;
    const __bf16* qr = Qb + ((size_t)((b * SS + qg) * NH + h)) * HD + lane * 4;
    float q0 = (float)qr[0], q1 = (float)qr[1], q2 = (float)qr[2], q3 = (float)qr[3];
#pragma unroll
    for (int j = 0; j < 3; j++) {
      const float* kr = Ksuf + ((((size_t)(b * NKV + hk) * LCK + j) * SS) + qg) * HD
                             + lane * 4;
      float part = q0 * kr[0] + q1 * kr[1] + q2 * kr[2] + q3 * kr[3];
      for (int off = 16; off; off >>= 1) part += __shfl_xor(part, off, 32);
      if ((rr >> 3) == hf) sdot[rr & 7][j] = part;
    }
  }
#pragma unroll
  for (int r = 0; r < 8; r++) {
    const int qg = qbase + r + 8 * hf;
    const bool qv = qg < valid;
    float e0 = qv ? sdot[r][0] * scale : NEGBIG;
    float e1 = qv ? sdot[r][1] * scale : NEGBIG;
    float e2 = qv ? sdot[r][2] * scale : NEGBIG;
    float mn = fmaxf(mrow[r], fmaxf(e0, fmaxf(e1, e2)));
    float al = __expf(mrow[r] - mn);
    mrow[r] = mn;
    float p0 = __expf(e0 - mn), p1 = __expf(e1 - mn), p2 = __expf(e2 - mn);
    lrow[r] = lrow[r] * al + p0 + p1 + p2;
    const size_t vb = (((size_t)(b * NKV + hk) * LCK) * SS + qg) * HD;
#pragma unroll
    for (int t = 0; t < 8; t++) {
      float a = p0 * Vsuf[vb + 0 * (size_t)SS * HD + t * 16 + n]
              + p1 * Vsuf[vb + 1 * (size_t)SS * HD + t * 16 + n]
              + p2 * Vsuf[vb + 2 * (size_t)SS * HD + t * 16 + n];
      Oacc[t][r] = Oacc[t][r] * al + a;
    }
  }

  // ---- normalize + write bf16 attention output [b,s,h,d] ----
#pragma unroll
  for (int r = 0; r < 8; r++) {
    const int qg = qbase + r + 8 * hf;
    float inv = (qg < valid) ? 1.0f / lrow[r] : 0.0f;
#pragma unroll
    for (int t = 0; t < 8; t++)
      Ob[((size_t)((b * SS + qg) * NH + h)) * HD + t * 16 + n] =
          (__bf16)(Oacc[t][r] * inv);
  }
}

// ------------------------------------------------------------------ driver ---
extern "C" void kernel_launch(void* const* d_in, const int* in_sizes, int n_in,
                              void* d_out, int out_size, void* d_ws, size_t ws_size,
                              hipStream_t stream) {
  const float* hidden = (const float*)d_in[0];
  const float* ksuf   = (const float*)d_in[1];
  const float* vsuf   = (const float*)d_in[2];
  const float* Wq     = (const float*)d_in[3];
  const float* Wk     = (const float*)d_in[4];
  const float* Wv     = (const float*)d_in[5];
  const float* Wo     = (const float*)d_in[6];
  const int*   validp = (const int*)d_in[7];
  float* out = (float*)d_out;

  char* ws = (char*)d_ws;
  size_t off = 0;
  auto carve = [&](size_t bytes) {
    char* p = ws + off;
    off = (off + bytes + 255) & ~(size_t)255;
    return p;
  };
  __bf16* Xbf  = (__bf16*)carve((size_t)MROWS * HIDDEN * 2);
  float*  Qf   = (float*) carve((size_t)MROWS * NH  * HD * 4);
  float*  Kf   = (float*) carve((size_t)MROWS * NKV * HD * 4);
  float*  Vf   = (float*) carve((size_t)MROWS * NKV * HD * 4);
  __bf16* Qbf  = (__bf16*)carve((size_t)MROWS * NH  * HD * 2);
  __bf16* Kbf  = (__bf16*)carve((size_t)MROWS * NKV * HD * 2);
  __bf16* Vtbf = (__bf16*)carve((size_t)MROWS * NKV * HD * 2);
  __bf16* Abf  = (__bf16*)carve((size_t)MROWS * NH  * HD * 2);

  // 1) hidden -> bf16
  {
    int n = MROWS * HIDDEN;
    f32_to_bf16_kernel<<<n / 256, 256, 0, stream>>>(hidden, Xbf, n);
  }
  // 2) projections
  gemm_bf16_kernel<<<dim3(NH * HD / GT_N, MROWS / GT_M), 256, 0, stream>>>(
      Xbf, Wq, Qf, MROWS, NH * HD, HIDDEN);
  gemm_bf16_kernel<<<dim3(NKV * HD / GT_N, MROWS / GT_M), 256, 0, stream>>>(
      Xbf, Wk, Kf, MROWS, NKV * HD, HIDDEN);
  gemm_bf16_kernel<<<dim3(NKV * HD / GT_N, MROWS / GT_M), 256, 0, stream>>>(
      Xbf, Wv, Vf, MROWS, NKV * HD, HIDDEN);
  // 3) RoPE -> bf16
  rope_bf16_kernel<<<(MROWS * NH * 64) / 256, 256, 0, stream>>>(Qf, Qbf, NH);
  rope_bf16_kernel<<<(MROWS * NKV * 64) / 256, 256, 0, stream>>>(Kf, Kbf, NKV);
  // 4) V transpose -> bf16
  vtrans_kernel<<<(MROWS * NKV * HD) / 256, 256, 0, stream>>>(Vf, Vtbf);
  // 5) flash attention
  attn_kernel<<<dim3(SS / 64, NH, BB), 128, 0, stream>>>(
      Qbf, Kbf, Vtbf, ksuf, vsuf, validp, Abf);
  // 6) output projection
  gemm_bf16_kernel<<<dim3(HIDDEN / GT_N, MROWS / GT_M), 256, 0, stream>>>(
      Abf, Wo, out, MROWS, HIDDEN, HIDDEN);
}
